// ViT_77988016160825
// MI455X (gfx1250) — matrix operand order repaired and checked by reading, not compiled, not run on hardware
//
#include <hip/hip_runtime.h>

// ---------------------------------------------------------------------------
// ViT-Base forward for MI455X (gfx1250, wave32, WMMA).
// All GEMMs -> v_wmma_f32_16x16x32_bf16. Flash-style attention (no 237MB score
// buffer). fp32 params converted to bf16 on the fly; fp32 accumulation.
// ---------------------------------------------------------------------------

typedef __attribute__((ext_vector_type(16))) __bf16 v16bf;
typedef __attribute__((ext_vector_type(8)))  __bf16 v8bf;
typedef __attribute__((ext_vector_type(8)))  float  v8f;

union AFrag { v16bf v; v8bf h[2]; __bf16 e[16]; };

#define WMMA_BF16(a,b,c) \
  __builtin_amdgcn_wmma_f32_16x16x32_bf16(false,(a),false,(b),(short)0,(c),false,false)

#define DMODEL 768
#define NHEAD  12
#define DHEAD  64
#define NTOK   785           // 1 + 28*28
#define BATCH  8
#define MTOK   (BATCH*NTOK)  // 6280 valid token rows
#define MPAD   6304          // padded to multiple of 32
#define MPATCH 6272          // B * 784 patches (already multiple of 32)
#define QTILES 50            // ceil(785/16)

// ---------------------------------------------------------------------------
// Generic bf16 WMMA GEMM:  C[M x Nn] = A[M x K] * Bt[Nn x K]^T + bias
// Wave computes a 32x32 tile: 2x2 grid of 16x16 WMMA accumulators.
// Operand fragments per the CDNA5 16-bit A-matrix layout:
//   A: lane m = l&15, element e -> k = e + (e>=8 ? 8 : 0) + 8*(l>=16)
//   B: lane n = l&15, element e -> k = e + (e>=8 ? 8 : 0) + 8*(l>=16)   (Bt is [n][k], k-contig)
// Epilogue modes: 0=fp32 store, 1=fp32 residual add, 2=bf16 store,
//                 3=patch-embed store (+pos embed, row remap), 4=GELU->bf16
// ---------------------------------------------------------------------------
__global__ __launch_bounds__(256)
void gemm_bf16_kernel(const __bf16* __restrict__ A, const __bf16* __restrict__ Bt,
                      const float* __restrict__ bias, float* __restrict__ outf,
                      __bf16* __restrict__ outb, const float* __restrict__ pos,
                      int K, int Nn, int Mvalid, int Ntile, int Tiles, int mode)
{
  int wave = threadIdx.x >> 5;
  int tile = blockIdx.x * 8 + wave;
  if (tile >= Tiles) return;
  int lane = threadIdx.x & 31;
  int half = lane >> 4, l15 = lane & 15;
  int ho = half * 8;     // A-fragment k sub-offset
  int bo = half * 16;    // unused; B loads also use ho
  int mt = tile / Ntile, nt = tile % Ntile;
  int m0 = mt * 32, n0 = nt * 32;

  v8f acc[2][2];
  for (int i = 0; i < 2; i++)
    for (int j = 0; j < 2; j++)
      for (int r = 0; r < 8; r++) acc[i][j][r] = 0.f;

  const __bf16* a0p = A  + (size_t)(m0 + l15)      * K;
  const __bf16* a1p = A  + (size_t)(m0 + 16 + l15) * K;
  const __bf16* b0p = Bt + (size_t)(n0 + l15)      * K;
  const __bf16* b1p = Bt + (size_t)(n0 + 16 + l15) * K;

  for (int k0 = 0; k0 < K; k0 += 32) {
    AFrag a0, a1, b0, b1;
    a0.h[0] = *(const v8bf*)(a0p + k0 + ho);
    a0.h[1] = *(const v8bf*)(a0p + k0 + 16 + ho);
    a1.h[0] = *(const v8bf*)(a1p + k0 + ho);
    a1.h[1] = *(const v8bf*)(a1p + k0 + 16 + ho);
    b0.h[0] = *(const v8bf*)(b0p + k0 + ho);
    b0.h[1] = *(const v8bf*)(b0p + k0 + 16 + ho);
    b1.h[0] = *(const v8bf*)(b1p + k0 + ho);
    b1.h[1] = *(const v8bf*)(b1p + k0 + 16 + ho);
    acc[0][0] = WMMA_BF16(a0.v, b0.v, acc[0][0]);
    acc[0][1] = WMMA_BF16(a0.v, b1.v, acc[0][1]);
    acc[1][0] = WMMA_BF16(a1.v, b0.v, acc[1][0]);
    acc[1][1] = WMMA_BF16(a1.v, b1.v, acc[1][1]);
  }

  for (int ms = 0; ms < 2; ms++)
    for (int ns = 0; ns < 2; ns++) {
      int gn = n0 + ns * 16 + l15;
      float bv = bias[gn];
      for (int r = 0; r < 8; r++) {
        int gm = m0 + ms * 16 + r + 8 * half;   // C layout: lane n=l&15, row r+8*(l>=16)
        float v = acc[ms][ns][r] + bv;
        if (mode == 0) {
          outf[(size_t)gm * Nn + gn] = v;
        } else if (mode == 1) {
          if (gm < Mvalid) outf[(size_t)gm * Nn + gn] += v;
        } else if (mode == 2) {
          outb[(size_t)gm * Nn + gn] = (__bf16)((gm < Mvalid) ? v : 0.f);
        } else if (mode == 3) {
          if (gm < Mvalid) {
            int bb = gm / 784, p = gm % 784;
            size_t row = (size_t)bb * NTOK + 1 + p;
            outf[row * DMODEL + gn] = v + pos[(size_t)(1 + p) * DMODEL + gn];
          }
        } else { // mode 4: exact GELU then bf16
          float g = 0.5f * v * (1.f + erff(v * 0.70710678118654752f));
          outb[(size_t)gm * Nn + gn] = (__bf16)((gm < Mvalid) ? g : 0.f);
        }
      }
    }
}

// ---------------------------------------------------------------------------
// Flash attention: 1 wave per (batch, head, 16-row Q tile); 32 keys/iter.
// S = Q*K^T via WMMA (contraction over dh=64 -> two k-steps), online softmax,
// P staged through LDS (accumulator layout -> A-operand layout), O += P*V.
// ---------------------------------------------------------------------------
__global__ __launch_bounds__(128)
void attn_kernel(const __bf16* __restrict__ qkv, __bf16* __restrict__ outb, float scale)
{
  __shared__ __bf16 lds[4][16 * 32] __attribute__((aligned(16)));
  int w    = threadIdx.x >> 5;
  int lane = threadIdx.x & 31;
  int g    = blockIdx.x * 4 + w;
  int qt   = g % QTILES;
  int hh   = (g / QTILES) % NHEAD;
  int bb   = g / (QTILES * NHEAD);
  int half = lane >> 4, l15 = lane & 15;
  int ho = half * 8, bo = half * 16;
  const size_t ldq = 3 * DMODEL;
  const __bf16* qbase = qkv + (size_t)bb * NTOK * ldq + (size_t)hh * DHEAD;
  const __bf16* kbase = qbase + DMODEL;
  const __bf16* vbase = qbase + 2 * DMODEL;

  // Q fragments (A layout), rows qt*16 + l15 (pad rows read zeros)
  AFrag aq[2];
  {
    const __bf16* qr = qbase + (size_t)(qt * 16 + l15) * ldq;
    for (int s = 0; s < 2; s++) {
      aq[s].h[0] = *(const v8bf*)(qr + s * 32 + ho);
      aq[s].h[1] = *(const v8bf*)(qr + s * 32 + 16 + ho);
    }
  }

  v8f acc[4];
  float mrow[8], lrow[8];
  for (int n = 0; n < 4; n++)
    for (int r = 0; r < 8; r++) acc[n][r] = 0.f;
  for (int r = 0; r < 8; r++) { mrow[r] = -1e30f; lrow[r] = 0.f; }

  for (int it = 0; it < 25; it++) {     // 25 * 32 = 800 >= 785 keys
    int key0 = it * 32;
    v8f s0, s1;
    for (int r = 0; r < 8; r++) { s0[r] = 0.f; s1[r] = 0.f; }

    // S tiles: keys [key0, key0+16) and [key0+16, key0+32)
    const __bf16* kr0 = kbase + (size_t)(key0 + l15)      * ldq;
    const __bf16* kr1 = kbase + (size_t)(key0 + 16 + l15) * ldq;
    for (int s = 0; s < 2; s++) {
      AFrag b0, b1;
      b0.h[0] = *(const v8bf*)(kr0 + s * 32 + ho);
      b0.h[1] = *(const v8bf*)(kr0 + s * 32 + 16 + ho);
      b1.h[0] = *(const v8bf*)(kr1 + s * 32 + ho);
      b1.h[1] = *(const v8bf*)(kr1 + s * 32 + 16 + ho);
      s0 = WMMA_BF16(aq[s].v, b0.v, s0);
      s1 = WMMA_BF16(aq[s].v, b1.v, s1);
    }

    // online softmax per output row (r + 8*half); cols strided across 16 lanes
    float alpha[8];
    for (int r = 0; r < 8; r++) {
      float a = s0[r] * scale; if (key0 + l15 >= NTOK)      a = -1e30f;
      float c = s1[r] * scale; if (key0 + 16 + l15 >= NTOK) c = -1e30f;
      float tmax = fmaxf(a, c);
      for (int o = 8; o > 0; o >>= 1) tmax = fmaxf(tmax, __shfl_xor(tmax, o, 32));
      float mnew = fmaxf(mrow[r], tmax);
      float al   = __expf(mrow[r] - mnew);
      float p0   = __expf(a - mnew);
      float p1   = __expf(c - mnew);
      float rs   = p0 + p1;
      for (int o = 8; o > 0; o >>= 1) rs += __shfl_xor(rs, o, 32);
      lrow[r] = lrow[r] * al + rs;
      mrow[r] = mnew;
      alpha[r] = al;
      int m = r + 8 * half;
      lds[w][m * 32 + l15]      = (__bf16)p0;
      lds[w][m * 32 + 16 + l15] = (__bf16)p1;
    }
    for (int n = 0; n < 4; n++)
      for (int r = 0; r < 8; r++) acc[n][r] *= alpha[r];
    __syncthreads();  // uniform trip count across all 4 waves

    // P as A operand (reload from LDS in A layout)
    AFrag pf;
    pf.h[0] = *(const v8bf*)(&lds[w][l15 * 32 + ho]);
    pf.h[1] = *(const v8bf*)(&lds[w][l15 * 32 + 16 + ho]);

    // V as B operand: n = dh col, k = key (strided gather)
    for (int n = 0; n < 4; n++) {
      AFrag vf;
      for (int e = 0; e < 16; e++) {
        int kk = (e & 7) + ho + ((e >> 3) << 4);   // B layout: k = (e&7) + 8*(l>=16) + 16*(e>=8)
        vf.e[e] = vbase[(size_t)(key0 + kk) * ldq + n * 16 + l15];
      }
      acc[n] = WMMA_BF16(pf.v, vf.v, acc[n]);
    }
    __syncthreads();
  }

  // epilogue: O = acc / l, concat heads, bf16 store (pad rows -> 0)
  for (int n = 0; n < 4; n++)
    for (int r = 0; r < 8; r++) {
      int t = qt * 16 + r + 8 * half;
      size_t row = (size_t)bb * NTOK + t;
      float o = (t < NTOK) ? acc[n][r] / lrow[r] : 0.f;
      outb[row * DMODEL + (size_t)hh * DHEAD + n * 16 + l15] = (__bf16)o;
    }
}

// ---------------------------------------------------------------------------
// LayerNorm (block per row, 768 = 3 * 256) -> bf16; pad rows forced to 0.
// ---------------------------------------------------------------------------
__device__ inline float block_reduce_sum(float v)
{
  for (int o = 16; o > 0; o >>= 1) v += __shfl_xor(v, o, 32);
  __shared__ float sh[8];
  int w = threadIdx.x >> 5;
  if ((threadIdx.x & 31) == 0) sh[w] = v;
  __syncthreads();
  float s = 0.f;
  for (int i = 0; i < 8; i++) s += sh[i];
  __syncthreads();
  return s;
}

__global__ __launch_bounds__(256)
void layernorm_kernel(const float* __restrict__ X, const float* __restrict__ g,
                      const float* __restrict__ b, __bf16* __restrict__ Y, int Mvalid)
{
  int row = blockIdx.x;
  const float* xr = X + (size_t)row * DMODEL;
  int c = threadIdx.x;
  float x0 = xr[c], x1 = xr[c + 256], x2 = xr[c + 512];
  float mu = block_reduce_sum(x0 + x1 + x2) * (1.f / 768.f);
  float d0 = x0 - mu, d1 = x1 - mu, d2 = x2 - mu;
  float var = block_reduce_sum(d0 * d0 + d1 * d1 + d2 * d2) * (1.f / 768.f);
  float rs = rsqrtf(var + 1e-5f);
  bool valid = row < Mvalid;
  __bf16* yr = Y + (size_t)row * DMODEL;
  yr[c]       = (__bf16)(valid ? d0 * rs * g[c]       + b[c]       : 0.f);
  yr[c + 256] = (__bf16)(valid ? d1 * rs * g[c + 256] + b[c + 256] : 0.f);
  yr[c + 512] = (__bf16)(valid ? d2 * rs * g[c + 512] + b[c + 512] : 0.f);
}

// Final LN of the CLS token rows only -> fp32 output [B, D]
__global__ __launch_bounds__(256)
void final_ln_kernel(const float* __restrict__ h, const float* __restrict__ g,
                     const float* __restrict__ b, float* __restrict__ out)
{
  int bb = blockIdx.x;
  const float* xr = h + (size_t)bb * NTOK * DMODEL;   // CLS row = token 0
  int c = threadIdx.x;
  float x0 = xr[c], x1 = xr[c + 256], x2 = xr[c + 512];
  float mu = block_reduce_sum(x0 + x1 + x2) * (1.f / 768.f);
  float d0 = x0 - mu, d1 = x1 - mu, d2 = x2 - mu;
  float var = block_reduce_sum(d0 * d0 + d1 * d1 + d2 * d2) * (1.f / 768.f);
  float rs = rsqrtf(var + 1e-5f);
  float* orow = out + (size_t)bb * DMODEL;
  orow[c]       = d0 * rs * g[c]       + b[c];
  orow[c + 256] = d1 * rs * g[c + 256] + b[c + 256];
  orow[c + 512] = d2 * rs * g[c + 512] + b[c + 512];
}

// ---------------------------------------------------------------------------
// Small prep kernels
// ---------------------------------------------------------------------------
__global__ __launch_bounds__(256)
void convert_kernel(const float* __restrict__ src, __bf16* __restrict__ dst, int n)
{
  int i = blockIdx.x * 256 + threadIdx.x;
  if (i < n) dst[i] = (__bf16)src[i];
}

// src [K][Nn] (in-major) -> dst [Nn][K] bf16 (out-major, k-contiguous)
__global__ __launch_bounds__(256)
void transpose_convert_kernel(const float* __restrict__ src, __bf16* __restrict__ dst,
                              int K, int Nn)
{
  int i = blockIdx.x * 256 + threadIdx.x;
  if (i >= K * Nn) return;
  int n = i / K, k = i % K;
  dst[i] = (__bf16)src[(size_t)k * Nn + n];
}

// im2col for 16x16 stride-16 patches: A[b*784+p][c*256+i*16+j] = x[b][c][...]
__global__ __launch_bounds__(256)
void im2col_kernel(const float* __restrict__ x, __bf16* __restrict__ A)
{
  int i = blockIdx.x * 256 + threadIdx.x;
  if (i >= MPATCH * DMODEL) return;
  int m = i / DMODEL, k = i % DMODEL;
  int bb = m / 784, p = m % 784;
  int py = p / 28, px = p % 28;
  int c = k >> 8, rem = k & 255, ii = rem >> 4, jj = rem & 15;
  A[i] = (__bf16)x[(((size_t)bb * 3 + c) * 448 + py * 16 + ii) * 448 + px * 16 + jj];
}

// CLS rows (h[b*785] = cls + pos[0]) + zero the pad rows 6280..6303
__global__ __launch_bounds__(256)
void cls_pad_kernel(const float* __restrict__ cls, const float* __restrict__ pos,
                    float* __restrict__ h)
{
  int i = blockIdx.x * 256 + threadIdx.x;
  if (i >= 32 * DMODEL) return;
  int rk = i / DMODEL, d = i % DMODEL;
  if (rk < BATCH) h[(size_t)rk * NTOK * DMODEL + d] = cls[d] + pos[d];
  else            h[(size_t)(MTOK + rk - BATCH) * DMODEL + d] = 0.f;
}

// ---------------------------------------------------------------------------
extern "C" void kernel_launch(void* const* d_in, const int* in_sizes, int n_in,
                              void* d_out, int out_size, void* d_ws, size_t ws_size,
                              hipStream_t stream)
{
  (void)in_sizes; (void)n_in; (void)out_size; (void)ws_size;
  const float* x       = (const float*)d_in[0];
  const float* conv_w  = (const float*)d_in[1];
  const float* conv_b  = (const float*)d_in[2];
  const float* cls_tok = (const float*)d_in[3];
  const float* pos_emb = (const float*)d_in[4];
  const float* ln1_g   = (const float*)d_in[5];
  const float* ln1_b   = (const float*)d_in[6];
  const float* qkv_w   = (const float*)d_in[7];
  const float* qkv_b   = (const float*)d_in[8];
  const float* proj_w  = (const float*)d_in[9];
  const float* proj_b  = (const float*)d_in[10];
  const float* ln2_g   = (const float*)d_in[11];
  const float* ln2_b   = (const float*)d_in[12];
  const float* mlp_w1  = (const float*)d_in[13];
  const float* mlp_b1  = (const float*)d_in[14];
  const float* mlp_w2  = (const float*)d_in[15];
  const float* mlp_b2  = (const float*)d_in[16];
  const float* lnf_g   = (const float*)d_in[17];
  const float* lnf_b   = (const float*)d_in[18];
  float* out = (float*)d_out;

  // workspace carve (256B aligned); total ~121 MB
  char* wp = (char*)d_ws;
  auto carve = [&](size_t bytes) -> void* {
    void* r = (void*)wp; wp += (bytes + 255) & ~(size_t)255; return r;
  };
  float*  h     = (float*) carve((size_t)MPAD * DMODEL * 4);   // residual stream
  __bf16* ybf   = (__bf16*)carve((size_t)MPAD * DMODEL * 2);   // LN output
  __bf16* qkvb  = (__bf16*)carve((size_t)MPAD * 3 * DMODEL * 2);
  __bf16* attnb = (__bf16*)carve((size_t)MPAD * DMODEL * 2);
  __bf16* ubf   = (__bf16*)carve((size_t)MPAD * 4 * DMODEL * 2);
  __bf16* apat  = (__bf16*)carve((size_t)MPATCH * DMODEL * 2);
  __bf16* wt    = (__bf16*)carve((size_t)3072 * 768 * 2);      // reused per GEMM

  // --- patch embed: h = im2col(x) @ conv_w^T + conv_b + pos ---
  { int n = 768 * 768;
    convert_kernel<<<(n + 255) / 256, 256, 0, stream>>>(conv_w, wt, n); }
  { int n = MPATCH * DMODEL;
    im2col_kernel<<<(n + 255) / 256, 256, 0, stream>>>(x, apat); }
  { int T = (MPATCH / 32) * (DMODEL / 32);
    gemm_bf16_kernel<<<(T + 7) / 8, 256, 0, stream>>>(
        apat, wt, conv_b, h, (__bf16*)nullptr, pos_emb,
        768, 768, MPATCH, DMODEL / 32, T, 3); }
  cls_pad_kernel<<<(32 * DMODEL + 255) / 256, 256, 0, stream>>>(cls_tok, pos_emb, h);

  // --- 12 transformer blocks ---
  for (int l = 0; l < 12; l++) {
    layernorm_kernel<<<MPAD, 256, 0, stream>>>(h, ln1_g + l * 768, ln1_b + l * 768,
                                               ybf, MTOK);
    { int n = 768 * 2304;
      transpose_convert_kernel<<<(n + 255) / 256, 256, 0, stream>>>(
          qkv_w + (size_t)l * n, wt, 768, 2304); }
    { int T = (MPAD / 32) * (2304 / 32);
      gemm_bf16_kernel<<<(T + 7) / 8, 256, 0, stream>>>(
          ybf, wt, qkv_b + l * 2304, (float*)nullptr, qkvb, pos_emb,
          768, 2304, MTOK, 2304 / 32, T, 2); }

    attn_kernel<<<(BATCH * NHEAD * QTILES) / 4, 128, 0, stream>>>(qkvb, attnb, 0.125f);

    { int n = 768 * 768;
      transpose_convert_kernel<<<(n + 255) / 256, 256, 0, stream>>>(
          proj_w + (size_t)l * n, wt, 768, 768); }
    { int T = (MPAD / 32) * (768 / 32);
      gemm_bf16_kernel<<<(T + 7) / 8, 256, 0, stream>>>(
          attnb, wt, proj_b + l * 768, h, (__bf16*)nullptr, pos_emb,
          768, 768, MTOK, 768 / 32, T, 1); }

    layernorm_kernel<<<MPAD, 256, 0, stream>>>(h, ln2_g + l * 768, ln2_b + l * 768,
                                               ybf, MTOK);
    { int n = 768 * 3072;
      transpose_convert_kernel<<<(n + 255) / 256, 256, 0, stream>>>(
          mlp_w1 + (size_t)l * n, wt, 768, 3072); }
    { int T = (MPAD / 32) * (3072 / 32);
      gemm_bf16_kernel<<<(T + 7) / 8, 256, 0, stream>>>(
          ybf, wt, mlp_b1 + l * 3072, (float*)nullptr, ubf, pos_emb,
          768, 3072, MTOK, 3072 / 32, T, 4); }
    { int n = 3072 * 768;
      transpose_convert_kernel<<<(n + 255) / 256, 256, 0, stream>>>(
          mlp_w2 + (size_t)l * n, wt, 3072, 768); }
    { int T = (MPAD / 32) * (768 / 32);
      gemm_bf16_kernel<<<(T + 7) / 8, 256, 0, stream>>>(
          ubf, wt, mlp_b2 + l * 768, h, (__bf16*)nullptr, pos_emb,
          3072, 768, MTOK, 768 / 32, T, 1); }
  }

  // --- final LN, CLS extraction ---
  final_ln_kernel<<<BATCH, 256, 0, stream>>>(h, lnf_g, lnf_b, out);
}
